// Model_Recursive_LSTM_v2_54288386621720
// MI455X (gfx1250) — matrix operand order repaired and checked by reading, not compiled, not run on hardware
//
#include <hip/hip_runtime.h>
#include <hip/hip_bf16.h>

// ---------------------------------------------------------------------------
// Model_Recursive_LSTM_v2 forward for MI455X (gfx1250, wave32, WMMA).
// All GEMMs: v_wmma_f32_16x16x32_bf16, 32x32 C-tile per wave, branch-free
// inner loop (all operands zero-padded to 32-aligned K/N in the workspace).
// ---------------------------------------------------------------------------

typedef __bf16 bf16;
typedef __attribute__((ext_vector_type(16))) __bf16 v16bf;
typedef __attribute__((ext_vector_type(8)))  __bf16 v8bf;
typedef __attribute__((ext_vector_type(8)))  float  v8f;

#define BB 512
#define NN 64
#define FF 1044
#define BN (BB * NN)          // 32768
#define HBI 100
#define EMBC 180
#define LDIM 20
#define NLOOPS 16

static inline int cdiv_h(int a, int b) { return (a + b - 1) / b; }
static inline int pad32(int x) { return (x + 31) & ~31; }

// ---------------- elementwise helpers ----------------
__global__ void k_vadd(float* __restrict__ d, const float* __restrict__ a,
                       const float* __restrict__ b, int n) {
  int i = blockIdx.x * blockDim.x + threadIdx.x;
  if (i < n) d[i] = a[i] + b[i];
}

__global__ void k_zero_bf16(bf16* __restrict__ p, int n) {
  int i = blockIdx.x * blockDim.x + threadIdx.x;
  if (i < n) p[i] = (bf16)0.f;
}

// weight f32[N,K] -> bf16[Np,Kp], zero padded
__global__ void k_pad_w(bf16* __restrict__ d, const float* __restrict__ s,
                        int N, int K, int Np, int Kp) {
  int i = blockIdx.x * blockDim.x + threadIdx.x;
  if (i >= Np * Kp) return;
  int r = i / Kp, c = i - r * Kp;
  d[i] = (r < N && c < K) ? (bf16)s[(long)r * K + c] : (bf16)0.f;
}

// rest = comps[:, 69:213] as (BN*4, 36) rows, padded to stride 64
__global__ void k_pack_rest(bf16* __restrict__ dst, const float* __restrict__ comps) {
  int i = blockIdx.x * blockDim.x + threadIdx.x;
  const int total = BN * 4 * 64;
  if (i >= total) return;
  int r = i >> 6, q = i & 63;          // r in [0, BN*4)
  int bn = r >> 2, t = r & 3;
  dst[i] = (q < 36) ? (bf16)comps[(long)bn * FF + 69 + t * 36 + q] : (bf16)0.f;
}

// x = [cols 0:69 of comps | hf(100) | hb(100) | comps cols 213:1044], stride 1120
__global__ void k_build_x(bf16* __restrict__ x, const float* __restrict__ comps,
                          const bf16* __restrict__ hf, const bf16* __restrict__ hb) {
  long i = (long)blockIdx.x * blockDim.x + threadIdx.x;
  const long total = (long)BN * 1120;
  if (i >= total) return;
  int r = (int)(i / 1120), c = (int)(i - (long)r * 1120);
  bf16 v;
  if      (c <   69) v = (bf16)comps[(long)r * FF + c];
  else if (c <  169) v = hf[r * 128 + (c - 69)];
  else if (c <  269) v = hb[r * 128 + (c - 169)];
  else if (c < 1100) v = (bf16)comps[(long)r * FF + (c - 56)];  // 269 -> col 213
  else               v = (bf16)0.f;
  x[i] = v;
}

// tree concat input: [nodes_h | comps_h | loops[:, li, :]] -> (512, 384) bf16
__global__ void k_build_cat(bf16* __restrict__ x,
                            const bf16* __restrict__ nodesH, const float* __restrict__ noNodes,
                            const bf16* __restrict__ compsH, const float* __restrict__ noComps,
                            const float* __restrict__ loops, int li) {
  int i = blockIdx.x * blockDim.x + threadIdx.x;
  const int total = BB * 384;
  if (i >= total) return;
  int b = i / 384, c = i - b * 384;
  bf16 v;
  if (c < EMBC)          v = nodesH ? nodesH[b * 192 + c]          : (bf16)noNodes[c];
  else if (c < 2 * EMBC) v = compsH ? compsH[b * 192 + (c - EMBC)] : (bf16)noComps[c - EMBC];
  else if (c < 380)      v = (bf16)loops[(long)b * NLOOPS * LDIM + li * LDIM + (c - 360)];
  else                   v = (bf16)0.f;
  x[i] = v;
}

// LSTM cell: gates [M,ldg] (i|f|g|o, each H wide), update c (compact), h (stride ldh)
__global__ void k_lstm_cell(const float* __restrict__ g4, int ldg,
                            float* __restrict__ c, bf16* __restrict__ h, int ldh,
                            int M, int H, int first) {
  int i = blockIdx.x * blockDim.x + threadIdx.x;
  if (i >= M * H) return;
  int m = i / H, j = i - m * H;
  const float* g = g4 + (long)m * ldg;
  float gi = g[j], gf = g[H + j], gg = g[2 * H + j], go = g[3 * H + j];
  float cprev = first ? 0.f : c[i];
  float si = 1.f / (1.f + __expf(-gi));
  float sf = 1.f / (1.f + __expf(-gf));
  float so = 1.f / (1.f + __expf(-go));
  float cn = sf * cprev + si * tanhf(gg);
  c[i] = cn;
  h[(long)m * ldh + j] = (bf16)(so * tanhf(cn));
}

// ---------------- WMMA GEMM ----------------
// C[M,N](f32, stride N) = act( A[M,Kp](bf16, stride lda) * W^T + bias [+ C] )
// W is bf16 [>=gridX*32, ldw], zero padded; Kp % 32 == 0; M % 32 == 0.
// Optional bf16 output Ch with stride ldch (pad columns written as 0).
// One wave computes a 32x32 C tile: 4 wmma per K step, branch-free loads.
// Operand layouts per CDNA5 ISA 7.12.2.
__global__ void __launch_bounds__(256)
k_gemm(const bf16* __restrict__ A, int lda,
       const bf16* __restrict__ W, int ldw,
       const float* __restrict__ bias,
       float* __restrict__ C, bf16* __restrict__ Ch, int ldch,
       int M, int N, int Kp, int beta, int act) {
  const int lane = threadIdx.x & 31;
  const int wave = threadIdx.x >> 5;
  const int m0 = (blockIdx.y * 8 + wave) * 32;
  const int n0 = blockIdx.x * 32;
  if (m0 >= M) return;

  // A fragment pointers: lane&15 = row-in-tile, (lane>>4)*8 = K sub-chunk
  const bf16* ap0 = A + (long)(m0 + (lane & 15)) * lda + ((lane >> 4) << 3);
  const bf16* ap1 = ap0 + (long)16 * lda;
  // B fragment pointers: lane&15 = W row (output col), (lane>>4)*8 = K sub-chunk
  const bf16* wp0 = W + (long)(n0 + (lane & 15)) * ldw + ((lane >> 4) << 3);
  const bf16* wp1 = wp0 + (long)16 * ldw;

  v8f acc00 = {}, acc01 = {}, acc10 = {}, acc11 = {};
  for (int k0 = 0; k0 < Kp; k0 += 32) {
    v8bf a0l = *(const v8bf*)(ap0 + k0);
    v8bf a0h = *(const v8bf*)(ap0 + k0 + 16);
    v8bf a1l = *(const v8bf*)(ap1 + k0);
    v8bf a1h = *(const v8bf*)(ap1 + k0 + 16);
    v8bf b0l = *(const v8bf*)(wp0 + k0);
    v8bf b0h = *(const v8bf*)(wp0 + k0 + 16);
    v8bf b1l = *(const v8bf*)(wp1 + k0);
    v8bf b1h = *(const v8bf*)(wp1 + k0 + 16);
    v16bf a0 = __builtin_shufflevector(a0l, a0h, 0,1,2,3,4,5,6,7,8,9,10,11,12,13,14,15);
    v16bf a1 = __builtin_shufflevector(a1l, a1h, 0,1,2,3,4,5,6,7,8,9,10,11,12,13,14,15);
    v16bf b0 = __builtin_shufflevector(b0l, b0h, 0,1,2,3,4,5,6,7,8,9,10,11,12,13,14,15);
    v16bf b1 = __builtin_shufflevector(b1l, b1h, 0,1,2,3,4,5,6,7,8,9,10,11,12,13,14,15);
    acc00 = __builtin_amdgcn_wmma_f32_16x16x32_bf16(false, a0, false, b0, (short)0, acc00, false, false);
    acc01 = __builtin_amdgcn_wmma_f32_16x16x32_bf16(false, a0, false, b1, (short)0, acc01, false, false);
    acc10 = __builtin_amdgcn_wmma_f32_16x16x32_bf16(false, a1, false, b0, (short)0, acc10, false, false);
    acc11 = __builtin_amdgcn_wmma_f32_16x16x32_bf16(false, a1, false, b1, (short)0, acc11, false, false);
  }

  const int colA = n0 + (lane & 15);
  const int colB = colA + 16;
  const int rb0 = m0 + 8 * (lane >> 4);
  const int rb1 = rb0 + 16;

  auto emit = [&](float av, int row, int col) {
    float v = 0.f;
    if (col < N) {
      v = av + (bias ? bias[col] : 0.f);
      if (beta) v += C[(long)row * N + col];
      if (act == 1)      v = (v > 0.f) ? v : (__expf(v) - 1.f);  // ELU
      else if (act == 2) v = fmaxf(v, 0.f);                      // ReLU
      C[(long)row * N + col] = v;
    }
    if (Ch) Ch[(long)row * ldch + col] = (bf16)v;
  };
#pragma unroll
  for (int r = 0; r < 8; r++) {
    emit(acc00[r], rb0 + r, colA);
    emit(acc01[r], rb0 + r, colB);
    emit(acc10[r], rb1 + r, colA);
    emit(acc11[r], rb1 + r, colB);
  }
}

// ---------------------------------------------------------------------------
extern "C" void kernel_launch(void* const* d_in, const int* in_sizes, int n_in,
                              void* d_out, int out_size, void* d_ws, size_t ws_size,
                              hipStream_t stream) {
  (void)in_sizes; (void)n_in; (void)out_size; (void)ws_size;

  // ---- inputs (params flattened as a jax pytree: dict keys sorted) ----
  const float* comps = (const float*)d_in[0];   // (512,64,1044)
  const float* loops = (const float*)d_in[1];   // (512,16,20)
  int ii = 2;
  const float* W0 = (const float*)d_in[ii++]; const float* b0 = (const float*)d_in[ii++];
  const float* W1 = (const float*)d_in[ii++]; const float* b1 = (const float*)d_in[ii++];
  const float* W2 = (const float*)d_in[ii++]; const float* b2 = (const float*)d_in[ii++];
  const float* W3 = (const float*)d_in[ii++]; const float* b3 = (const float*)d_in[ii++];
  const float* WihB = (const float*)d_in[ii++]; const float* WhhB = (const float*)d_in[ii++];
  const float* bihB = (const float*)d_in[ii++]; const float* bhhB = (const float*)d_in[ii++];
  const float* WihF = (const float*)d_in[ii++]; const float* WhhF = (const float*)d_in[ii++];
  const float* bihF = (const float*)d_in[ii++]; const float* bhhF = (const float*)d_in[ii++];
  const float* WihC = (const float*)d_in[ii++]; const float* WhhC = (const float*)d_in[ii++];
  const float* bihC = (const float*)d_in[ii++]; const float* bhhC = (const float*)d_in[ii++];
  const float* Wc0 = (const float*)d_in[ii++]; const float* bc0 = (const float*)d_in[ii++];
  const float* Wc1 = (const float*)d_in[ii++]; const float* bc1 = (const float*)d_in[ii++];
  const float* We  = (const float*)d_in[ii++]; const float* be  = (const float*)d_in[ii++];
  const float* noC = (const float*)d_in[ii++];
  const float* noN = (const float*)d_in[ii++];
  const float* WihN = (const float*)d_in[ii++]; const float* WhhN = (const float*)d_in[ii++];
  const float* bihN = (const float*)d_in[ii++]; const float* bhhN = (const float*)d_in[ii++];
  const float* Wp  = (const float*)d_in[ii++]; const float* bp  = (const float*)d_in[ii++];
  const float* Wr0 = (const float*)d_in[ii++]; const float* br0 = (const float*)d_in[ii++];
  const float* Wr1 = (const float*)d_in[ii++]; const float* br1 = (const float*)d_in[ii++];

  // ---- workspace allocator ----
  char* wsb = (char*)d_ws;
  size_t off = 0;
  auto alloc = [&](size_t bytes) -> void* {
    void* p = wsb + off;
    off = (off + bytes + 255) & ~(size_t)255;
    return p;
  };

  auto padw = [&](const float* src, int Nr, int Kc) -> bf16* {
    int Np = pad32(Nr), Kp = pad32(Kc);
    bf16* d = (bf16*)alloc((size_t)Np * Kp * sizeof(bf16));
    k_pad_w<<<cdiv_h(Np * Kp, 256), 256, 0, stream>>>(d, src, Nr, Kc, Np, Kp);
    return d;
  };
  auto combine = [&](const float* a, const float* b, int n) -> float* {
    float* d = (float*)alloc((size_t)n * sizeof(float));
    k_vadd<<<cdiv_h(n, 256), 256, 0, stream>>>(d, a, b, n);
    return d;
  };

  // ---- padded bf16 weights (small; L2-resident) ----
  bf16* We_b   = padw(We, 36, 36);                                   // 64x64
  bf16* WihF_b = padw(WihF, 400, 36);  bf16* WhhF_b = padw(WhhF, 400, 100);
  bf16* WihB_b = padw(WihB, 400, 36);  bf16* WhhB_b = padw(WhhB, 400, 100);
  bf16* W0_b = padw(W0, 600, 1100);    bf16* W1_b = padw(W1, 350, 600);
  bf16* W2_b = padw(W2, 200, 350);     bf16* W3_b = padw(W3, 180, 200);
  bf16* WihC_b = padw(WihC, 720, 180); bf16* WhhC_b = padw(WhhC, 720, 180);
  bf16* WihN_b = padw(WihN, 720, 180); bf16* WhhN_b = padw(WhhN, 720, 180);
  bf16* Wc0_b = padw(Wc0, 200, 380);   bf16* Wc1_b = padw(Wc1, 180, 200);
  bf16* Wr0_b = padw(Wr0, 200, 180);   bf16* Wr1_b = padw(Wr1, 180, 200);
  bf16* Wp_b  = padw(Wp, 1, 180);                                    // 32x192
  float* bcF = combine(bihF, bhhF, 400);
  float* bcB = combine(bihB, bhhB, 400);
  float* bcC = combine(bihC, bhhC, 720);
  float* bcN = combine(bihN, bhhN, 720);

  // ---- activations (all bf16 strides are pad32(width)) ----
  bf16*  enc   = (bf16*)alloc((size_t)BN * 4 * 64 * sizeof(bf16));   // (BN*4, 64)
  float* Cf32  = (float*)alloc((size_t)BN * 600 * sizeof(float));    // f32 C / gates
  float* cbuf  = (float*)alloc((size_t)BN * HBI * sizeof(float));    // cell state (compact)
  bf16*  hf    = (bf16*)alloc((size_t)BN * 128 * sizeof(bf16));
  bf16*  hb    = (bf16*)alloc((size_t)BN * 128 * sizeof(bf16));
  bf16*  ping  = (bf16*)alloc((size_t)BN * 1120 * sizeof(bf16));     // rest / x / a1
  bf16*  pong  = (bf16*)alloc((size_t)BN * 608 * sizeof(bf16));      // a0 / a2
  bf16*  cemb  = (bf16*)alloc((size_t)BN * 192 * sizeof(bf16));      // comps_emb
  bf16*  nodeH[7];
  for (int n = 0; n < 7; n++) nodeH[n] = (bf16*)alloc((size_t)BB * 192 * sizeof(bf16));
  bf16* nState = (bf16*)alloc((size_t)BB * 192 * sizeof(bf16));
  bf16* cState = (bf16*)alloc((size_t)BB * 192 * sizeof(bf16));
  bf16* xcat   = (bf16*)alloc((size_t)BB * 384 * sizeof(bf16));
  bf16* t200   = (bf16*)alloc((size_t)BB * 224 * sizeof(bf16));
  bf16* r180   = (bf16*)alloc((size_t)BB * 192 * sizeof(bf16));

  // zero h-state buffers once (their pad columns must read as 0 in GEMMs)
  k_zero_bf16<<<cdiv_h(BN * 128, 256), 256, 0, stream>>>(hf, BN * 128);
  k_zero_bf16<<<cdiv_h(BN * 128, 256), 256, 0, stream>>>(hb, BN * 128);
  k_zero_bf16<<<cdiv_h(BB * 192, 256), 256, 0, stream>>>(nState, BB * 192);
  k_zero_bf16<<<cdiv_h(BB * 192, 256), 256, 0, stream>>>(cState, BB * 192);

  auto gemm = [&](const bf16* A, int lda, const bf16* W, int ldw, int Kp,
                  const float* bias, float* C, bf16* Ch, int ldch,
                  int M, int Nn, int beta, int act) {
    int gx = Ch ? (ldch / 32) : cdiv_h(Nn, 32);
    dim3 g((unsigned)gx, (unsigned)cdiv_h(M, 256));
    k_gemm<<<g, 256, 0, stream>>>(A, lda, W, ldw, bias, C, Ch, ldch, M, Nn, Kp, beta, act);
  };
  auto lstm_step = [&](const bf16* xA, int xlda, int xKp, const bf16* Wih,
                       const bf16* Whh, const float* bc, bf16* h, int ldh,
                       float* c, int M, int H, int t) {
    gemm(xA, xlda, Wih, xKp, xKp, bc, Cf32, nullptr, 0, M, 4 * H, 0, 0);
    if (t > 0) gemm(h, ldh, Whh, ldh, ldh, nullptr, Cf32, nullptr, 0, M, 4 * H, 1, 0);
    k_lstm_cell<<<cdiv_h(M * H, 256), 256, 0, stream>>>(Cf32, 4 * H, c, h, ldh, M, H, t == 0);
  };

  // ---- stage A: encode vectors (BN*4 x 36 @ 36x36) ----
  k_pack_rest<<<cdiv_h(BN * 4 * 64, 256), 256, 0, stream>>>(ping, comps);
  gemm(ping, 64, We_b, 64, 64, be, Cf32, enc, 64, BN * 4, 36, 0, 0);

  // ---- stage B: bidirectional LSTM (H=100, T=4) ----
  for (int t = 0; t < 4; t++)
    lstm_step(enc + t * 64, 256, 64, WihF_b, WhhF_b, bcF, hf, 128, cbuf, BN, HBI, t);
  for (int s = 0; s < 4; s++)   // reversed sequence
    lstm_step(enc + (3 - s) * 64, 256, 64, WihB_b, WhhB_b, bcB, hb, 128, cbuf, BN, HBI, s);

  // ---- stage C/D: comp-embed MLP (ELU) ----
  k_build_x<<<(int)(((long)BN * 1120 + 255) / 256), 256, 0, stream>>>(ping, comps, hf, hb);
  gemm(ping, 1120, W0_b, 1120, 1120, b0, Cf32, pong, 608, BN, 600, 0, 1);
  gemm(pong, 608,  W1_b, 608,  608,  b1, Cf32, ping, 352, BN, 350, 0, 1);
  gemm(ping, 352,  W2_b, 352,  352,  b2, Cf32, pong, 224, BN, 200, 0, 1);
  gemm(pong, 224,  W3_b, 224,  224,  b3, Cf32, cemb, 192, BN, EMBC, 0, 1);

  // ---- stage E: static tree (post-order) ----
  struct TNode { int li, cb, ce, nch, ch[3]; };
  const TNode nd[7] = {
      {0, -1, -1, 3, {1, 2, 3}},
      {1,  0, 16, 2, {4, 5, 0}},
      {2, 32, 48, 0, {0, 0, 0}},
      {3, 48, 64, 1, {6, 0, 0}},
      {4, 16, 24, 0, {0, 0, 0}},
      {5, 24, 32, 0, {0, 0, 0}},
      {6, 48, 56, 0, {0, 0, 0}},
  };
  const int order[7] = {4, 5, 1, 2, 6, 3, 0};

  for (int oi = 0; oi < 7; oi++) {
    const TNode& nnd = nd[order[oi]];
    bf16* nh = nullptr;
    if (nnd.nch > 0) {  // nodes_lstm over children hidden states
      for (int t = 0; t < nnd.nch; t++)
        lstm_step(nodeH[nnd.ch[t]], 192, 192, WihN_b, WhhN_b, bcN,
                  nState, 192, cbuf, BB, EMBC, t);
      nh = nState;
    }
    bf16* chp = nullptr;
    if (nnd.cb >= 0) {  // comps_lstm over selected comps_emb rows
      for (int t = 0; t < nnd.ce - nnd.cb; t++)
        lstm_step(cemb + (long)(nnd.cb + t) * 192, NN * 192, 192,
                  WihC_b, WhhC_b, bcC, cState, 192, cbuf, BB, EMBC, t);
      chp = cState;
    }
    k_build_cat<<<cdiv_h(BB * 384, 256), 256, 0, stream>>>(xcat, nh, noN, chp, noC,
                                                           loops, nnd.li);
    gemm(xcat, 384, Wc0_b, 384, 384, bc0, Cf32, t200, 224, BB, 200, 0, 1);
    gemm(t200, 224, Wc1_b, 224, 224, bc1, Cf32, nodeH[order[oi]], 192, BB, EMBC, 0, 1);
  }

  // ---- stage F: regression + predict ----
  gemm(nodeH[0], 192, Wr0_b, 192, 192, br0, Cf32, t200, 224, BB, 200, 0, 1);
  gemm(t200, 224, Wr1_b, 224, 224, br1, Cf32, r180, 192, BB, EMBC, 0, 1);
  gemm(r180, 192, Wp_b, 192, 192, bp, (float*)d_out, nullptr, 0, BB, 1, 0, 2);  // ReLU
}
